// MoE_81612968558627
// MI455X (gfx1250) — matrix operand order repaired and checked by reading, not compiled, not run on hardware
//
#include <hip/hip_runtime.h>
#include <hip/hip_bf16.h>

#define TT 4096   // B*S tokens
#define DD 1024
#define FF 512
#define EE 8
#define RR 64
#define HH 256

typedef __attribute__((ext_vector_type(16))) _Float16 v16h;
typedef __attribute__((ext_vector_type(8)))  _Float16 v8h;
typedef __attribute__((ext_vector_type(8)))  float    v8f;

// ---------------------------------------------------------------- utilities

__global__ void zero_f32_kernel(float* __restrict__ p, int n) {
    int i = blockIdx.x * 256 + threadIdx.x;
    if (i < n) p[i] = 0.0f;
}

__global__ void cvt_f16_kernel(const float* __restrict__ in, _Float16* __restrict__ out, int n) {
    int i = blockIdx.x * 256 + threadIdx.x;
    if (i < n) out[i] = (_Float16)in[i];
}

// per-tensor max|x| reduction into a single float slot (slot pre-zeroed;
// non-negative floats compare correctly as uint bit patterns)
__global__ void maxabs_kernel(const float* __restrict__ x, int n, float* __restrict__ slot) {
    __shared__ float red[256];
    float m = 0.0f;
    for (int i = blockIdx.x * 256 + threadIdx.x; i < n; i += gridDim.x * 256)
        m = fmaxf(m, fabsf(x[i]));
    red[threadIdx.x] = m;
    __syncthreads();
    for (int s = 128; s > 0; s >>= 1) {
        if (threadIdx.x < s) red[threadIdx.x] = fmaxf(red[threadIdx.x], red[threadIdx.x + s]);
        __syncthreads();
    }
    if (threadIdx.x == 0)
        atomicMax((unsigned int*)slot, __float_as_uint(red[0]));
}

__device__ __forceinline__ float fq_apply(float x, float maxabs) {
    float s = fmaxf(maxabs * (1.0f / 127.0f), 1e-8f);
    float q = rintf(x / s);                    // round-half-even, like jnp.round
    q = fminf(fmaxf(q, -128.0f), 127.0f);
    return q * s;
}

__global__ void quantize_kernel(float* __restrict__ x, int n, const float* __restrict__ slot) {
    int i = blockIdx.x * 256 + threadIdx.x;
    if (i < n) x[i] = fq_apply(x[i], *slot);
}

// in: [Nr][Kc] f32  ->  out: [Kc][Nr] f16, optional per-tensor fake-quant
__global__ void transpose_cvt_kernel(const float* __restrict__ in, _Float16* __restrict__ out,
                                     int Nr, int Kc, const float* __restrict__ slot) {
    int total = Nr * Kc;
    float ma = slot ? *slot : 0.0f;
    for (int idx = blockIdx.x * 256 + threadIdx.x; idx < total; idx += gridDim.x * 256) {
        int k = idx / Nr;
        int n = idx - k * Nr;
        float v = in[(size_t)n * Kc + k];
        if (slot) v = fq_apply(v, ma);
        out[idx] = (_Float16)v;
    }
}

// h = fq(y1) * sigmoid(fq(y3))  (both already quantized in place), emit f16
__global__ void gate_fuse_kernel(const float* __restrict__ y1, const float* __restrict__ y3,
                                 _Float16* __restrict__ h, int n) {
    int i = blockIdx.x * 256 + threadIdx.x;
    if (i < n) {
        float g = 1.0f / (1.0f + expf(-y3[i]));
        h[i] = (_Float16)(y1[i] * g);
    }
}

// out[t,d] += cw[t,e] * y2[t,d]
__global__ void combine_kernel(float* __restrict__ out, const float* __restrict__ y2,
                               const float* __restrict__ cw, int e, int n) {
    int i = blockIdx.x * 256 + threadIdx.x;
    if (i < n) {
        int t = i / DD;
        out[i] += cw[t * EE + e] * y2[i];
    }
}

__global__ void add_kernel(float* __restrict__ out, const float* __restrict__ z, int n) {
    int i = blockIdx.x * 256 + threadIdx.x;
    if (i < n) out[i] += z[i];
}

// ---------------------------------------------------------------- WMMA GEMM
// C[M,N] (+)= A[M,K] * Bt[K,N], f16 inputs, f32 accumulate/out.
// 128x64 block tile, 8 waves (wave32) in a 4x2 layout: each wave owns a
// 32x32 patch = 4 x (16x16) WMMA tiles, K stepped by 32. LDS is double
// buffered: next k-tile is fetched to registers during compute, stored to the
// alternate buffer, one workgroup barrier per k-step.
__global__ __launch_bounds__(256)
void gemm_f16_wmma_kernel(const _Float16* __restrict__ A, const _Float16* __restrict__ Bt,
                          float* __restrict__ C, int M, int N, int Kd, int accumulate) {
    __shared__ __align__(16) _Float16 As[2][128][32];
    __shared__ __align__(16) _Float16 Bs[2][32][64];

    const int tid  = threadIdx.x;
    const int lane = tid & 31;
    const int wave = tid >> 5;
    const int wm   = wave & 3;   // 0..3 -> 32-row group
    const int wn   = wave >> 2;  // 0..1 -> 32-col half
    const int m0   = blockIdx.y * 128;
    const int n0   = blockIdx.x * 64;

    v8f acc00 = {}, acc01 = {}, acc10 = {}, acc11 = {};

    // cooperative tile loads: A 128x32 (16 halves/thread), B 32x64 (8/thread)
    const int ar = tid >> 1, ac = (tid & 1) * 16;
    const int br = tid >> 3, bc = (tid & 7) * 8;
    const _Float16* aRow = A + (size_t)(m0 + ar) * Kd + ac;
    const _Float16* bRow = Bt + (size_t)br * N + n0 + bc;

    // fragment addresses (per wave/lane)
    const int mrow0 = wm * 32 + (lane & 15);        // first 16-row subtile
    const int hk    = (lane >> 4) * 8;              // K-half select for A frag
    const int bn    = wn * 32;                      // N base for B frags

    // preload k-tile 0 into buffer 0
    {
        *(v8h*)(&As[0][ar][ac])     = *(const v8h*)(aRow);
        *(v8h*)(&As[0][ar][ac + 8]) = *(const v8h*)(aRow + 8);
        *(v8h*)(&Bs[0][br][bc])     = *(const v8h*)(bRow);
    }
    __syncthreads();

    int buf = 0;
    for (int k0 = 0; k0 < Kd; k0 += 32) {
        const int nxt = buf ^ 1;
        const bool haveNext = (k0 + 32) < Kd;
        v8h a0n = {}, a1n = {}, bnx = {};
        if (haveNext) {
            a0n = *(const v8h*)(aRow + k0 + 32);
            a1n = *(const v8h*)(aRow + k0 + 40);
            bnx = *(const v8h*)(bRow + (size_t)(k0 + 32) * N);
        }
        if (k0 + 64 < Kd) {
            __builtin_prefetch((const void*)(aRow + k0 + 64), 0, 1);
            __builtin_prefetch((const void*)(bRow + (size_t)(k0 + 64) * N), 0, 1);
        }

        // build fragments from current buffer and issue 4 WMMAs
        union { v16h v; v8h h[2]; } af0, af1, bf0, bf1;
        af0.h[0] = *(const v8h*)(&As[buf][mrow0][hk]);
        af0.h[1] = *(const v8h*)(&As[buf][mrow0][hk + 16]);
        af1.h[0] = *(const v8h*)(&As[buf][mrow0 + 16][hk]);
        af1.h[1] = *(const v8h*)(&As[buf][mrow0 + 16][hk + 16]);
        for (int s = 0; s < 16; ++s) {
            const int kk = hk + (s < 8 ? s : 8 + s);
            bf0.v[s] = Bs[buf][kk][bn + (lane & 15)];
            bf1.v[s] = Bs[buf][kk][bn + 16 + (lane & 15)];
        }

        acc00 = __builtin_amdgcn_wmma_f32_16x16x32_f16(false, af0.v, false, bf0.v,
                                                       (short)0, acc00, false, false);
        acc01 = __builtin_amdgcn_wmma_f32_16x16x32_f16(false, af0.v, false, bf1.v,
                                                       (short)0, acc01, false, false);
        acc10 = __builtin_amdgcn_wmma_f32_16x16x32_f16(false, af1.v, false, bf0.v,
                                                       (short)0, acc10, false, false);
        acc11 = __builtin_amdgcn_wmma_f32_16x16x32_f16(false, af1.v, false, bf1.v,
                                                       (short)0, acc11, false, false);

        // stage next k-tile into the alternate buffer
        if (haveNext) {
            *(v8h*)(&As[nxt][ar][ac])     = a0n;
            *(v8h*)(&As[nxt][ar][ac + 8]) = a1n;
            *(v8h*)(&Bs[nxt][br][bc])     = bnx;
        }
        __syncthreads();
        buf = nxt;
    }

    // C/D layout: lanes 0-15: N=lane, M=vgpr r; lanes 16-31: N=lane-16, M=8+r
    const int nl    = lane & 15;
    const int rbase = (lane >> 4) << 3;
    const int col0  = n0 + wn * 32 + nl;
    for (int r = 0; r < 8; ++r) {
        const int row0 = m0 + wm * 32 + rbase + r;        // first M subtile
        const int row1 = row0 + 16;                       // second M subtile
        size_t i00 = (size_t)row0 * N + col0;
        size_t i10 = (size_t)row1 * N + col0;
        if (accumulate) {
            C[i00]      += acc00[r];
            C[i00 + 16] += acc01[r];
            C[i10]      += acc10[r];
            C[i10 + 16] += acc11[r];
        } else {
            C[i00]      = acc00[r];
            C[i00 + 16] = acc01[r];
            C[i10]      = acc10[r];
            C[i10 + 16] = acc11[r];
        }
    }
}

// ---------------------------------------------------------------- router
__global__ __launch_bounds__(256)
void router_kernel(const float* __restrict__ xf,
                   const float* __restrict__ scorer_w, const float* __restrict__ scorer_b,
                   const float* __restrict__ p1_w, const float* __restrict__ p1_b,
                   const float* __restrict__ p2_w, const float* __restrict__ p2_b,
                   float* __restrict__ cw, float* __restrict__ usage,
                   float* __restrict__ ent_sum, float* __restrict__ z_sum) {
    __shared__ float xs[DD];
    __shared__ float hid[HH];
    __shared__ float base[EE];
    __shared__ float pol[EE];
    const int t = blockIdx.x;
    const float* xrow = xf + (size_t)t * DD;
    for (int i = threadIdx.x; i < DD; i += 256) xs[i] = xrow[i];
    __syncthreads();
    {   // policy hidden: one output per thread (HH==256)
        const float* w = p1_w + (size_t)threadIdx.x * DD;
        float acc = p1_b[threadIdx.x];
        for (int j = 0; j < DD; ++j) acc = fmaf(xs[j], w[j], acc);
        hid[threadIdx.x] = fmaxf(acc, 0.0f);
    }
    __syncthreads();
    if (threadIdx.x < EE) {
        const int e = threadIdx.x;
        const float* w = scorer_w + (size_t)e * DD;
        float acc = scorer_b[e];
        for (int j = 0; j < DD; ++j) acc = fmaf(xs[j], w[j], acc);
        base[e] = acc;                              // TEMP == 1.0
        const float* w2 = p2_w + (size_t)e * HH;
        float acc2 = p2_b[e];
        for (int j = 0; j < HH; ++j) acc2 = fmaf(hid[j], w2[j], acc2);
        pol[e] = acc2;
    }
    __syncthreads();
    if (threadIdx.x == 0) {
        float pm = pol[0], bm = base[0];
        for (int e = 1; e < EE; ++e) { pm = fmaxf(pm, pol[e]); bm = fmaxf(bm, base[e]); }
        float pe[EE], ps = 0.0f, bs = 0.0f;
        for (int e = 0; e < EE; ++e) { pe[e] = expf(pol[e] - pm); ps += pe[e]; bs += expf(base[e] - bm); }
        float lse = bm + logf(bs);                  // logsumexp(base)
        float lg[EE], lm = -3.4e38f;
        for (int e = 0; e < EE; ++e) { lg[e] = 0.5f * (base[e] + pe[e] / ps); lm = fmaxf(lm, lg[e]); }
        float sc[EE], ss = 0.0f;
        for (int e = 0; e < EE; ++e) { sc[e] = expf(lg[e] - lm); ss += sc[e]; }
        float ent = 0.0f;
        for (int e = 0; e < EE; ++e) { sc[e] /= ss; ent -= sc[e] * logf(sc[e] + 1e-6f); }
        // top-2 (ties -> lowest index, like lax.top_k)
        int i1 = 0;
        for (int e = 1; e < EE; ++e) if (sc[e] > sc[i1]) i1 = e;
        int i2 = (i1 == 0) ? 1 : 0;
        for (int e = 0; e < EE; ++e) if (e != i1 && sc[e] > sc[i2]) i2 = e;
        float* cwrow = cw + (size_t)t * EE;
        for (int e = 0; e < EE; ++e) cwrow[e] = 0.0f;
        cwrow[i1] = sc[i1];
        cwrow[i2] = sc[i2];
        atomicAdd(&usage[i1], sc[i1]);
        atomicAdd(&usage[i2], sc[i2]);
        atomicAdd(ent_sum, ent);
        atomicAdd(z_sum, lse * lse);
    }
}

__global__ void finalize_kernel(const float* __restrict__ slots, float* __restrict__ tail) {
    if (threadIdx.x == 0 && blockIdx.x == 0) {
        float usum = 0.0f;
        for (int e = 0; e < EE; ++e) { tail[e] = slots[e]; usum += slots[e]; }
        float mean = usum / (float)EE;
        float uf[EE], ufm = 0.0f;
        for (int e = 0; e < EE; ++e) { uf[e] = slots[e] / (mean + 1e-6f); ufm += uf[e]; }
        ufm /= (float)EE;
        float var = 0.0f;
        for (int e = 0; e < EE; ++e) { float d = uf[e] - ufm; var += d * d; }
        var /= (float)(EE - 1);
        float ent_mean = slots[8] / (float)TT;
        float z_loss   = 1e-4f * (slots[9] / (float)TT);
        tail[EE]     = 0.3f * var - 0.1f * ent_mean + z_loss;
        tail[EE + 1] = 0.0f;
        tail[EE + 2] = 0.0f;
    }
}

// ---------------------------------------------------------------- host side

extern "C" void kernel_launch(void* const* d_in, const int* in_sizes, int n_in,
                              void* d_out, int out_size, void* d_ws, size_t ws_size,
                              hipStream_t stream) {
    (void)in_sizes; (void)n_in; (void)out_size;
    const float* x        = (const float*)d_in[0];
    const float* scorer_w = (const float*)d_in[1];
    const float* scorer_b = (const float*)d_in[2];
    const float* p1_w     = (const float*)d_in[3];
    const float* p1_b     = (const float*)d_in[4];
    const float* p2_w     = (const float*)d_in[5];
    const float* p2_b     = (const float*)d_in[6];
    const float* ew1_w    = (const float*)d_in[7];
    const float* ew1_a    = (const float*)d_in[8];
    const float* ew1_b    = (const float*)d_in[9];
    const float* ew2_w    = (const float*)d_in[10];
    const float* ew2_a    = (const float*)d_in[11];
    const float* ew2_b    = (const float*)d_in[12];
    const float* ew3_w    = (const float*)d_in[13];
    const float* ew3_a    = (const float*)d_in[14];
    const float* ew3_b    = (const float*)d_in[15];
    const float* s1_w = (const float*)d_in[16];
    const float* s1_a = (const float*)d_in[17];
    const float* s1_b = (const float*)d_in[18];
    const float* s2_w = (const float*)d_in[19];
    const float* s2_a = (const float*)d_in[20];
    const float* s2_b = (const float*)d_in[21];
    const float* s3_w = (const float*)d_in[22];
    const float* s3_a = (const float*)d_in[23];
    const float* s3_b = (const float*)d_in[24];
    float* out = (float*)d_out;

    // ---- workspace carve-out (256B aligned bump allocator)
    char* base = (char*)d_ws;
    size_t off = 0;
    auto alloc = [&](size_t bytes) -> void* {
        off = (off + 255) & ~(size_t)255;
        void* p = base + off;
        off += bytes;
        return p;
    };
    float*     slots = (float*)    alloc(512 * sizeof(float));
    float*     cw    = (float*)    alloc((size_t)TT * EE * 4);
    _Float16*  xf16  = (_Float16*) alloc((size_t)TT * DD * 2);
    float*     y1    = (float*)    alloc((size_t)TT * FF * 4);
    float*     y3    = (float*)    alloc((size_t)TT * FF * 4);
    float*     y2    = (float*)    alloc((size_t)TT * DD * 4);
    float*     u     = (float*)    alloc((size_t)TT * RR * 4);
    _Float16*  u16   = (_Float16*) alloc((size_t)TT * RR * 2);
    _Float16*  h16   = (_Float16*) alloc((size_t)TT * FF * 2);
    _Float16*  w1t   = (_Float16*) alloc((size_t)DD * FF * 2);
    _Float16*  w3t   = (_Float16*) alloc((size_t)DD * FF * 2);
    _Float16*  w2t   = (_Float16*) alloc((size_t)FF * DD * 2);
    _Float16*  b1t   = (_Float16*) alloc((size_t)DD * RR * 2);
    _Float16*  a1t   = (_Float16*) alloc((size_t)RR * FF * 2);
    _Float16*  b3t   = (_Float16*) alloc((size_t)DD * RR * 2);
    _Float16*  a3t   = (_Float16*) alloc((size_t)RR * FF * 2);
    _Float16*  b2t   = (_Float16*) alloc((size_t)FF * RR * 2);
    _Float16*  a2t   = (_Float16*) alloc((size_t)RR * DD * 2);
    if (off > ws_size) return;   // workspace too small -> refuse to scribble

    int slot_idx = 10;           // 0..7 usage, 8 entropy sum, 9 z sum

    auto zero = [&](float* p, size_t n) {
        zero_f32_kernel<<<(unsigned)((n + 255) / 256), 256, 0, stream>>>(p, (int)n);
    };
    auto cvt = [&](const float* in, _Float16* o, size_t n) {
        cvt_f16_kernel<<<(unsigned)((n + 255) / 256), 256, 0, stream>>>(in, o, (int)n);
    };
    auto gemm = [&](const _Float16* A, const _Float16* Bt, float* C,
                    int M, int N, int Kd, int acc) {
        dim3 g((unsigned)(N / 64), (unsigned)(M / 128));
        gemm_f16_wmma_kernel<<<g, 256, 0, stream>>>(A, Bt, C, M, N, Kd, acc);
    };
    auto fq_inplace = [&](float* p, size_t n) {
        float* s = slots + (slot_idx++);
        maxabs_kernel<<<256, 256, 0, stream>>>(p, (int)n, s);
        quantize_kernel<<<(unsigned)((n + 255) / 256), 256, 0, stream>>>(p, (int)n, s);
    };
    auto prep_w = [&](const float* w, _Float16* wt, int Nr, int Kc, bool fq) {
        const float* s = nullptr;
        if (fq) {
            float* sp = slots + (slot_idx++);
            maxabs_kernel<<<256, 256, 0, stream>>>(w, Nr * Kc, sp);
            s = sp;
        }
        transpose_cvt_kernel<<<(unsigned)(((size_t)Nr * Kc + 255) / 256), 256, 0, stream>>>(
            w, wt, Nr, Kc, s);
    };
    auto run_ffn = [&](const float* w1, const float* a1, const float* b1,
                       const float* w2, const float* a2, const float* b2,
                       const float* w3, const float* a3, const float* b3) {
        prep_w(w1, w1t, FF, DD, true);
        prep_w(w3, w3t, FF, DD, true);
        prep_w(w2, w2t, DD, FF, true);
        prep_w(b1, b1t, RR, DD, false);
        prep_w(a1, a1t, FF, RR, false);
        prep_w(b3, b3t, RR, DD, false);
        prep_w(a3, a3t, FF, RR, false);
        prep_w(b2, b2t, RR, FF, false);
        prep_w(a2, a2t, DD, RR, false);

        // y1 = x @ fq(w1)^T + (x @ b1^T) @ a1^T ; then fq
        gemm(xf16, w1t, y1, TT, FF, DD, 0);
        gemm(xf16, b1t, u,  TT, RR, DD, 0);
        cvt(u, u16, (size_t)TT * RR);
        gemm(u16, a1t, y1, TT, FF, RR, 1);
        fq_inplace(y1, (size_t)TT * FF);

        // y3 (gate logits), same shape; then fq
        gemm(xf16, w3t, y3, TT, FF, DD, 0);
        gemm(xf16, b3t, u,  TT, RR, DD, 0);
        cvt(u, u16, (size_t)TT * RR);
        gemm(u16, a3t, y3, TT, FF, RR, 1);
        fq_inplace(y3, (size_t)TT * FF);

        // h = fq(y1) * sigmoid(fq(y3))
        gate_fuse_kernel<<<(unsigned)(((size_t)TT * FF + 255) / 256), 256, 0, stream>>>(
            y1, y3, h16, TT * FF);

        // y2 = h @ fq(w2)^T + (h @ b2^T) @ a2^T ; fq twice (lora_lin fq + ffn fq)
        gemm(h16, w2t, y2, TT, DD, FF, 0);
        gemm(h16, b2t, u,  TT, RR, FF, 0);
        cvt(u, u16, (size_t)TT * RR);
        gemm(u16, a2t, y2, TT, DD, RR, 1);
        fq_inplace(y2, (size_t)TT * DD);
        fq_inplace(y2, (size_t)TT * DD);
    };

    // ---- schedule
    zero(slots, 512);
    zero(out, (size_t)TT * DD);
    cvt(x, xf16, (size_t)TT * DD);

    router_kernel<<<TT, 256, 0, stream>>>(x, scorer_w, scorer_b, p1_w, p1_b, p2_w, p2_b,
                                          cw, slots, slots + 8, slots + 9);

    for (int e = 0; e < EE; ++e) {
        run_ffn(ew1_w + (size_t)e * FF * DD, ew1_a + (size_t)e * FF * RR, ew1_b + (size_t)e * RR * DD,
                ew2_w + (size_t)e * DD * FF, ew2_a + (size_t)e * DD * RR, ew2_b + (size_t)e * RR * FF,
                ew3_w + (size_t)e * FF * DD, ew3_a + (size_t)e * FF * RR, ew3_b + (size_t)e * RR * DD);
        combine_kernel<<<(unsigned)(((size_t)TT * DD + 255) / 256), 256, 0, stream>>>(
            out, y2, cw, e, TT * DD);
    }

    // shared expert, added densely
    run_ffn(s1_w, s1_a, s1_b, s2_w, s2_a, s2_b, s3_w, s3_a, s3_b);
    add_kernel<<<(unsigned)(((size_t)TT * DD + 255) / 256), 256, 0, stream>>>(out, y2, TT * DD);

    finalize_kernel<<<1, 1, 0, stream>>>(slots, out + (size_t)TT * DD);
}
